// HRQuantizeEMAReset_22582938043137
// MI455X (gfx1250) — matrix operand order repaired and checked by reading, not compiled, not run on hardware
//
#include <hip/hip_runtime.h>
#include <hip/hip_bf16.h>

typedef __attribute__((ext_vector_type(16))) __bf16 v16bf;
typedef __attribute__((ext_vector_type(8)))  float  v8f;

#define N_  64
#define C_  512
#define T_  512
#define K_  1024
#define NCT (64*512*512)
#define LDSTRIDE 520   // 512 + 8 pad; rows 16B-aligned, reduces LDS bank conflicts
#define RPB 32         // rows per block in argmin (two 16-row A tiles)

__device__ __forceinline__ unsigned short f2bf(float f) {
    unsigned int u = __float_as_uint(f);
    u += 0x7FFFu + ((u >> 16) & 1u);            // round-to-nearest-even
    return (unsigned short)(u >> 16);
}

// map float to uint s.t. uint order == float order (ascending)
__device__ __forceinline__ unsigned int ford(float f) {
    unsigned int u = __float_as_uint(f);
    return (u & 0x80000000u) ? ~u : (u | 0x80000000u);
}

__device__ __forceinline__ unsigned long long packKey(float score, int entry) {
    return ((unsigned long long)ford(score) << 32) | (unsigned int)entry;
}

// ---------------- init: residual = x, f_hat = 0, zero counts/loss ----------------
__global__ void k_init(const float* __restrict__ x, float* __restrict__ residual,
                       float* __restrict__ fhat, float* __restrict__ counts,
                       float* __restrict__ lossSum) {
    size_t i = (size_t)blockIdx.x * blockDim.x + threadIdx.x;
    if (i < (size_t)NCT) { residual[i] = x[i]; fhat[i] = 0.f; }
    if (i < K_) counts[i] = 0.f;
    if (i == 0) lossSum[0] = 0.f;
}

// ---------------- codebook -> bf16, and ||c||^2 ----------------
__global__ void k_cbprep(const float* __restrict__ cb, unsigned short* __restrict__ cbbf,
                         float* __restrict__ cbsq) {
    int k = blockIdx.x, tid = threadIdx.x;
    __shared__ float sh[256];
    float s = 0.f;
    for (int c = tid; c < C_; c += 256) {
        float v = cb[(size_t)k * C_ + c];
        cbbf[(size_t)k * C_ + c] = f2bf(v);
        s += v * v;
    }
    sh[tid] = s; __syncthreads();
    for (int st = 128; st > 0; st >>= 1) { if (tid < st) sh[tid] += sh[tid + st]; __syncthreads(); }
    if (tid == 0) cbsq[k] = sh[0];
}

// ---------------- masked average-pool -> bf16 rows (N*t, C) row-major ----------------
__global__ void k_down(const float* __restrict__ residual, const int* __restrict__ mlen,
                       unsigned short* __restrict__ rdbf, int scale, int t) {
    size_t i = (size_t)blockIdx.x * blockDim.x + threadIdx.x;
    size_t total = (size_t)N_ * t * C_;
    if (i >= total) return;
    int c    = (int)(i % C_);
    int row  = (int)(i / C_);
    int time = row % t;
    int n    = row / t;
    int ml   = mlen[n];
    const float* rp = residual + ((size_t)n * C_ + c) * T_ + (size_t)time * scale;
    float s = 0.f;
    for (int j = 0; j < scale; ++j) {
        int tau = time * scale + j;
        s += (tau < ml) ? rp[j] : 0.f;
    }
    rdbf[(size_t)row * C_ + c] = f2bf(s / (float)scale);
}

// ---------------- WMMA nearest-codebook argmin ----------------
// One block = 32 rows (two 16-row A tiles in LDS). 8 waves; wave w handles
// column-tile PAIRS w, w+8, w+16, w+24 (32 pairs cover K=1024 entries).
// 2x2 register blocking: per k-chunk, 2 LDS A loads + 2 global B loads -> 4 WMMAs.
// score = ||c||^2 - 2*dot (||v||^2 is row-constant -> irrelevant for argmin).
__global__ __launch_bounds__(256) void k_argmin(const unsigned short* __restrict__ rdbf,
                                                const unsigned short* __restrict__ cbbf,
                                                const float* __restrict__ cbsq,
                                                int* __restrict__ idxOut) {
    __shared__ unsigned short ldsA[RPB * LDSTRIDE];          // ~33.3 KB
    __shared__ unsigned long long rowMin[RPB];
    int tid = threadIdx.x;
    int rowBase = blockIdx.x * RPB;

    // cooperative load of 32 x 512 bf16 A tiles (rows contiguous in rdbf)
    {
        int r  = tid >> 3;       // 0..31 rows
        int ch = tid & 7;        // 8 chunks of 64 ushorts (128B) per row
        const uint4* src = (const uint4*)(rdbf + ((size_t)(rowBase + r)) * C_ + ch * 64);
        uint4* dst = (uint4*)(&ldsA[r * LDSTRIDE + ch * 64]);
#pragma unroll
        for (int q = 0; q < 8; ++q) dst[q] = src[q];
    }
    if (tid < RPB) rowMin[tid] = ~0ull;
    __syncthreads();

    int lane = tid & 31;
    int wave = tid >> 5;
    int half = lane >> 4;     // K-group selector for A/B fragments
    int nn   = lane & 15;     // A-row within tile; also B/D column

    unsigned long long best0[8], best1[8];   // per D-row-slot keys, row tiles 0 and 1
#pragma unroll
    for (int r = 0; r < 8; ++r) { best0[r] = ~0ull; best1[r] = ~0ull; }

    union UA { v16bf bf; uint4 q[2]; };

    for (int cp = wave; cp < K_ / 32; cp += 8) {             // 4 pairs per wave
        int e0 = cp * 32 + nn;                               // first col tile entry
        int e1 = e0 + 16;                                    // second col tile entry
        const unsigned short* b0row = cbbf + (size_t)e0 * C_;
        const unsigned short* b1row = cbbf + (size_t)e1 * C_;
        if (cp + 8 < K_ / 32) {                              // prefetch next pair's rows
            __builtin_prefetch(cbbf + (size_t)(e0 + 256) * C_, 0, 1);
            __builtin_prefetch(cbbf + (size_t)(e1 + 256) * C_, 0, 1);
        }
        v8f acc00 = {}, acc01 = {}, acc10 = {}, acc11 = {};
#pragma unroll 2
        for (int kc = 0; kc < C_; kc += 32) {
            UA a0, a1, b0, b1;
            // A 16x32 bf16 lane layout (ISA 7.12.2): two contiguous 16B runs at
            // k = kc + 8*half and k = kc + 16 + 8*half for row nn (tile0) / nn+16 (tile1)
            const unsigned short* ar0 = &ldsA[nn * LDSTRIDE + kc + 8 * half];
            const unsigned short* ar1 = &ldsA[(nn + 16) * LDSTRIDE + kc + 8 * half];
            a0.q[0] = *(const uint4*)(ar0);      a0.q[1] = *(const uint4*)(ar0 + 16);
            a1.q[0] = *(const uint4*)(ar1);      a1.q[1] = *(const uint4*)(ar1 + 16);
            // B 32x16 bf16: lane holds its column's two 16B runs at kc + 8*half and kc + 16 + 8*half
            const uint4* bp0 = (const uint4*)(b0row + kc + 8 * half);
            const uint4* bp1 = (const uint4*)(b1row + kc + 8 * half);
            b0.q[0] = bp0[0]; b0.q[1] = bp0[2];
            b1.q[0] = bp1[0]; b1.q[1] = bp1[2];
            acc00 = __builtin_amdgcn_wmma_f32_16x16x32_bf16(false, a0.bf, false, b0.bf,
                                                            (short)0, acc00, false, false);
            acc01 = __builtin_amdgcn_wmma_f32_16x16x32_bf16(false, a0.bf, false, b1.bf,
                                                            (short)0, acc01, false, false);
            acc10 = __builtin_amdgcn_wmma_f32_16x16x32_bf16(false, a1.bf, false, b0.bf,
                                                            (short)0, acc10, false, false);
            acc11 = __builtin_amdgcn_wmma_f32_16x16x32_bf16(false, a1.bf, false, b1.bf,
                                                            (short)0, acc11, false, false);
        }
        float cq0 = cbsq[e0];
        float cq1 = cbsq[e1];
#pragma unroll
        for (int r = 0; r < 8; ++r) {
            // D layout: vgpr r, lane -> row m = r + 8*half, col n = nn
            unsigned long long k00 = packKey(__builtin_fmaf(-2.0f, acc00[r], cq0), e0);
            unsigned long long k01 = packKey(__builtin_fmaf(-2.0f, acc01[r], cq1), e1);
            unsigned long long k10 = packKey(__builtin_fmaf(-2.0f, acc10[r], cq0), e0);
            unsigned long long k11 = packKey(__builtin_fmaf(-2.0f, acc11[r], cq1), e1);
            unsigned long long m0 = k00 < k01 ? k00 : k01;
            unsigned long long m1 = k10 < k11 ? k10 : k11;
            best0[r] = m0 < best0[r] ? m0 : best0[r];
            best1[r] = m1 < best1[r] ? m1 : best1[r];
        }
    }
#pragma unroll
    for (int r = 0; r < 8; ++r) {
        int mloc = r + 8 * half;
        atomicMin(&rowMin[mloc],      best0[r]);   // ds_min_u64
        atomicMin(&rowMin[16 + mloc], best1[r]);
    }
    __syncthreads();
    if (tid < RPB) idxOut[rowBase + tid] = (int)(rowMin[tid] & 0xFFFFFFFFull);
}

// ---------------- fused gather + linear upsample + residual/f_hat update ----------------
__global__ void k_up(const float* __restrict__ cb, const int* __restrict__ idx,
                     const int* __restrict__ mlen, float* __restrict__ residual,
                     float* __restrict__ fhat, int scale, int t) {
    size_t i = (size_t)blockIdx.x * blockDim.x + threadIdx.x;
    if (i >= (size_t)NCT) return;
    int tau = (int)(i % T_);
    int nc  = (int)(i / T_);
    int n = nc / C_;
    int c = nc % C_;
    float pos  = ((float)tau + 0.5f) / (float)scale - 0.5f;
    float pmax = (float)(t - 1);
    pos = pos < 0.f ? 0.f : (pos > pmax ? pmax : pos);
    int lo = (int)pos;                       // floor (pos >= 0)
    int hi = (lo + 1 < t) ? lo + 1 : t - 1;
    float w = pos - (float)lo;
    int tlim = mlen[n] / scale;
    int base = n * t;
    float v0 = (lo < tlim) ? cb[(size_t)idx[base + lo] * C_ + c] : 0.f;
    float v1 = (hi < tlim) ? cb[(size_t)idx[base + hi] * C_ + c] : 0.f;
    float up = v0 * (1.f - w) + v1 * w;
    residual[i] -= up;
    fhat[i]     += up;
}

// ---------------- masked MSE partial sum ----------------
__global__ void k_loss(const float* __restrict__ x, const float* __restrict__ fhat,
                       const int* __restrict__ mlen, float* __restrict__ lossSum) {
    __shared__ float sh[256];
    size_t i = (size_t)blockIdx.x * blockDim.x + threadIdx.x;
    float v = 0.f;
    if (i < (size_t)NCT) {
        int tau = (int)(i % T_);
        int n   = (int)(i / ((size_t)C_ * T_));
        if (tau < mlen[n]) { float d = x[i] - fhat[i]; v = d * d; }
    }
    int tid = threadIdx.x;
    sh[tid] = v; __syncthreads();
    for (int s = 128; s > 0; s >>= 1) { if (tid < s) sh[tid] += sh[tid + s]; __syncthreads(); }
    if (tid == 0) atomicAdd(lossSum, sh[0]);
}

// ---------------- masked histogram ----------------
__global__ void k_hist(const int* __restrict__ idx, const int* __restrict__ mlen,
                       float* __restrict__ counts, int scale, int t) {
    int r = blockIdx.x * blockDim.x + threadIdx.x;
    if (r >= N_ * t) return;
    int n = r / t, time = r % t;
    if (time < mlen[n] / scale) atomicAdd(&counts[idx[r]], 1.0f);
}

// ---------------- finalize: loss, perplexity ----------------
__global__ void k_final(const int* __restrict__ mlen, const float* __restrict__ counts,
                        const float* __restrict__ lossSum, float* __restrict__ outTail) {
    __shared__ float sh[256];
    __shared__ float s_ms, s_tot;
    int tid = threadIdx.x;
    float v = 0.f;
    for (int i = tid; i < N_; i += 256) v += (float)mlen[i];
    sh[tid] = v; __syncthreads();
    for (int s = 128; s > 0; s >>= 1) { if (tid < s) sh[tid] += sh[tid + s]; __syncthreads(); }
    if (tid == 0) s_ms = sh[0];
    __syncthreads();
    v = 0.f;
    for (int i = tid; i < K_; i += 256) v += counts[i];
    sh[tid] = v; __syncthreads();
    for (int s = 128; s > 0; s >>= 1) { if (tid < s) sh[tid] += sh[tid + s]; __syncthreads(); }
    if (tid == 0) s_tot = sh[0];
    __syncthreads();
    v = 0.f;
    for (int i = tid; i < K_; i += 256) { float p = counts[i] / s_tot; v += p * logf(p + 1e-7f); }
    sh[tid] = v; __syncthreads();
    for (int s = 128; s > 0; s >>= 1) { if (tid < s) sh[tid] += sh[tid + s]; __syncthreads(); }
    if (tid == 0) {
        outTail[0] = lossSum[0] / (s_ms * (float)T_) / 3.0f;   // loss
        outTail[1] = expf(-sh[0]);                             // perplexity
    }
}

extern "C" void kernel_launch(void* const* d_in, const int* in_sizes, int n_in,
                              void* d_out, int out_size, void* d_ws, size_t ws_size,
                              hipStream_t stream) {
    const float* x   = (const float*)d_in[0];
    const float* cb  = (const float*)d_in[1];
    const int* mlen  = (const int*)d_in[2];
    float* out  = (float*)d_out;
    float* fhat = out;                        // f_hat accumulates directly into d_out
    float* outTail = out + (size_t)NCT;       // [loss, perplexity]

    char* w = (char*)d_ws;
    size_t off = 0;
    float*          residual = (float*)(w + off);          off += (size_t)NCT * 4;
    unsigned short* rdbf     = (unsigned short*)(w + off); off += (size_t)N_ * T_ * C_ * 2;
    unsigned short* cbbf     = (unsigned short*)(w + off); off += (size_t)K_ * C_ * 2;
    float*          cbsq     = (float*)(w + off);          off += (size_t)K_ * 4;
    int*            idxbuf   = (int*)(w + off);            off += (size_t)N_ * T_ * 4;
    float*          counts   = (float*)(w + off);          off += (size_t)K_ * 4;
    float*          lossSum  = (float*)(w + off);          off += 64;

    const int threads = 256;
    const int gNCT = (NCT + threads - 1) / threads;

    k_init<<<gNCT, threads, 0, stream>>>(x, residual, fhat, counts, lossSum);
    k_cbprep<<<K_, threads, 0, stream>>>(cb, cbbf, cbsq);

    const int scales[3] = {4, 2, 1};
    for (int si = 0; si < 3; ++si) {
        int s = scales[si], t = T_ / s, R = N_ * t;
        k_down<<<(int)(((size_t)R * C_ + threads - 1) / threads), threads, 0, stream>>>(
            residual, mlen, rdbf, s, t);
        k_argmin<<<R / RPB, threads, 0, stream>>>(rdbf, cbbf, cbsq, idxbuf);
        k_up<<<gNCT, threads, 0, stream>>>(cb, idxbuf, mlen, residual, fhat, s, t);
        k_loss<<<gNCT, threads, 0, stream>>>(x, fhat, mlen, lossSum);
        k_hist<<<(R + threads - 1) / threads, threads, 0, stream>>>(idxbuf, mlen, counts, s, t);
    }
    k_final<<<1, threads, 0, stream>>>(mlen, counts, lossSum, outTail);

    (void)in_sizes; (void)n_in; (void)out_size; (void)ws_size;
}
